// RNN_6751688590197
// MI455X (gfx1250) — matrix operand order repaired and checked by reading, not compiled, not run on hardware
//
#include <hip/hip_runtime.h>

// ---------------------------------------------------------------------------
// Elman RNN for MI455X (gfx1250, wave32).
//   xp = x @ W_ih^T + b_ih + b_hh            (throughput GEMM -> kernel 1)
//   h_t = tanh(xp_t + h_{t-1} @ W_hh^T)      (latency-bound scan -> kernel 2)
//
// Roofline: 268 MB mandatory HBM traffic (~11.5us @ 23.3 TB/s); compute ~34
// GFLOP (trivial at WMMA rates).  The limiter is the 1024-step dependency
// chain -> persistent-RNN scan: batch rows are independent, so B=128 splits
// into 8 workgroups with NO inter-WG sync; W_hh^T lives in registers as
// pinned f16 WMMA B-fragments; h ping-pongs between two LDS buffers with a
// single workgroup barrier per step.  xp (128 MiB) is staged in d_ws and
// fits in the 192 MB L2, so the scan re-reads it at L2 bandwidth.
// ---------------------------------------------------------------------------

typedef _Float16 v8h  __attribute__((ext_vector_type(8)));
typedef _Float16 v16h __attribute__((ext_vector_type(16)));
typedef float    v8f  __attribute__((ext_vector_type(8)));

#define T_STEPS 1024
#define BATCH   128
#define HDIM    256
#define LDM     (HDIM + 8)   // padded f16 LDS row (528 B) -> bank skew, 16B aligned

#if __has_builtin(__builtin_amdgcn_tanhf)
  #define FAST_TANH(x) __builtin_amdgcn_tanhf(x)   // v_tanh_f32
#elif __has_builtin(__builtin_amdgcn_tanh_f32)
  #define FAST_TANH(x) __builtin_amdgcn_tanh_f32(x)
#else
  #define FAST_TANH(x) tanhf(x)
#endif

// A-fragment (16x32 f16): lane(0-15)=M (lanes16-31 repeat M, upper-K half).
// VGPR j<4 : K = kb*32 + half*8 + {2j, 2j+1};  VGPR j>=4 : K += 16.
// => two contiguous 16-byte LDS reads per lane (ds_load_b128 x2).
static __device__ __forceinline__ v16h load_a_frag(const _Float16* row, int kb, int half) {
  const v8h lo = *(const v8h*)(row + kb * 32 + half * 8);
  const v8h hi = *(const v8h*)(row + kb * 32 + 16 + half * 8);
  return __builtin_shufflevector(lo, hi, 0,1,2,3,4,5,6,7,8,9,10,11,12,13,14,15);
}

// B-fragment (32x16 f16): lane(0-15)=N, same K order as A-fragment:
// VGPR j<4 : K = half*8 + {2j, 2j+1};  VGPR j>=4 : K += 8.  Our B = W^T, so B[k][n] = W[n][k]:
// each lane reads consecutive fp32 pairs from row N of W and converts.
// The empty asm pins the converted f16 tuple in VGPRs: without it the
// compiler keeps the fp32 sources alive and re-runs v_cvt_pk_f16_f32 at
// every use (observed in round-1 asm) - fatal inside the 1024-step loop.
static __device__ __forceinline__ v16h make_b_frag(const float* wrow, int kb, int half) {
  v16h b;
  #pragma unroll
  for (int j = 0; j < 8; ++j) {
    const float2 w = *(const float2*)(wrow + kb * 32 + half * 8 + 2 * j + (j >= 4 ? 8 : 0));
    b[2 * j]     = (_Float16)w.x;
    b[2 * j + 1] = (_Float16)w.y;
  }
  asm volatile("" : "+v"(b));   // opaque: hold f16 fragment, drop fp32 sources
  return b;
}

static __device__ __forceinline__ v8f wmma16(v16h a, v16h b, v8f c) {
  return __builtin_amdgcn_wmma_f32_16x16x32_f16(false, a, false, b, (short)0, c,
                                                false, false);
}

// ---------------------------------------------------------------------------
// Kernel 1: xp[T*B, H] = x[T*B, K] @ W_ih^T + (b_ih + b_hh)
// 2048 blocks x 512 threads (16 waves).  Block owns 64 rows x all 256 cols.
// x block staged to LDS as f16; each wave: 1 column tile, 4 M-tiles,
// W_ih B-fragments register-resident (64 VGPRs), 8 WMMAs per tile.
// ---------------------------------------------------------------------------
__global__ __launch_bounds__(512) void rnn_xproj(const float* __restrict__ x,
                                                 const float* __restrict__ W_ih,
                                                 const float* __restrict__ b_ih,
                                                 const float* __restrict__ b_hh,
                                                 float* __restrict__ xp) {
  __shared__ _Float16 at[64 * LDM];

  const int tid  = threadIdx.x;
  const int wave = tid >> 5;          // 0..15 -> N tile
  const int lane = tid & 31;
  const int nl   = lane & 15;
  const int half = lane >> 4;
  const size_t row0 = (size_t)blockIdx.x * 64;

  // Cooperative stage: 64x256 fp32 -> f16 LDS (float4 loads, coalesced).
  #pragma unroll
  for (int i = 0; i < 8; ++i) {
    const int e4 = tid + i * 512;          // 4096 float4 in the block
    const int r  = e4 >> 6;                // 64 float4 per row
    const int c4 = (e4 & 63) << 2;
    const float4 v = *(const float4*)(x + (row0 + r) * HDIM + c4);
    _Float16* d = &at[r * LDM + c4];
    d[0] = (_Float16)v.x; d[1] = (_Float16)v.y;
    d[2] = (_Float16)v.z; d[3] = (_Float16)v.w;
  }

  // Register-resident B-fragments of W_ih^T for this wave's column tile.
  const int col = wave * 16 + nl;
  const float* wrow = W_ih + (size_t)col * HDIM;
  v16h bfrag[8];
  #pragma unroll
  for (int kb = 0; kb < 8; ++kb) bfrag[kb] = make_b_frag(wrow, kb, half);

  __syncthreads();

  v8f acc[4] = {};
  #pragma unroll
  for (int kb = 0; kb < 8; ++kb) {
    #pragma unroll
    for (int mt = 0; mt < 4; ++mt) {
      const v16h a = load_a_frag(&at[(mt * 16 + nl) * LDM], kb, half);
      acc[mt] = wmma16(a, bfrag[kb], acc[mt]);
    }
  }

  const float bias = b_ih[col] + b_hh[col];
  #pragma unroll
  for (int mt = 0; mt < 4; ++mt) {
    float* orow = xp + (row0 + mt * 16 + 8 * half) * HDIM + col;  // M = v + 8*half
    #pragma unroll
    for (int v = 0; v < 8; ++v) orow[(size_t)v * HDIM] = acc[mt][v] + bias;
  }
}

// ---------------------------------------------------------------------------
// Kernel 2: persistent recurrence.  8 blocks (16 batch rows each, independent)
// x 512 threads (16 waves; wave owns one 16-col tile of H).  W_hh^T fragments
// stay pinned in registers for all 1024 steps; h_{t-1} is read from LDS as
// WMMA A-fragments, h_t written to the other LDS buffer (ping-pong) so ONE
// barrier per step suffices: all reads of buf[p] and writes of buf[1-p]
// complete before the barrier, so step t+1 can safely read buf[1-p] and
// overwrite buf[p].  Per-step critical path: 16 ds_load_b128 -> two 4-deep
// WMMA chains -> v_tanh -> barrier.  xp loads are recurrence-independent
// (next step prefetched via global_prefetch).
// ---------------------------------------------------------------------------
__global__ __launch_bounds__(512) void rnn_recur(const float* __restrict__ xp,
                                                 const float* __restrict__ W_hh,
                                                 float* __restrict__ out) {
  __shared__ _Float16 hbuf[2][16 * LDM];

  const int tid   = threadIdx.x;
  const int wave  = tid >> 5;         // N tile
  const int lane  = tid & 31;
  const int nl    = lane & 15;
  const int half  = lane >> 4;
  const int brow0 = blockIdx.x * 16;  // this block's batch slice
  const int col   = wave * 16 + nl;

  // h_0 = 0 in buffer 0
  for (int i = tid; i < 16 * LDM; i += 512) hbuf[0][i] = (_Float16)0.f;

  // Persistent, pinned B-fragments of W_hh^T (64 VGPRs/lane, loaded once).
  const float* wrow = W_hh + (size_t)col * HDIM;
  v16h bfrag[8];
  #pragma unroll
  for (int kb = 0; kb < 8; ++kb) bfrag[kb] = make_b_frag(wrow, kb, half);

  __syncthreads();

  for (int t = 0; t < T_STEPS; ++t) {
    const _Float16* arow = &hbuf[t & 1][nl * LDM];        // read h_{t-1}
    _Float16* hnext      = &hbuf[(t & 1) ^ 1][0];         // write h_t

    // xp tile for this step: independent of the recurrence, issue first.
    const float* xpt = xp + ((size_t)t * BATCH + brow0 + 8 * half) * HDIM + col;
    float xv[8];
    #pragma unroll
    for (int v = 0; v < 8; ++v) xv[v] = xpt[(size_t)v * HDIM];  // M = v + 8*half
    if (t + 1 < T_STEPS)
      __builtin_prefetch(xp + ((size_t)(t + 1) * BATCH + brow0 + 8 * half) * HDIM + col,
                         0, 0);  // global_prefetch for next step

    // h_{t-1} @ W_hh^T : two interleaved 4-deep accumulation chains.
    v8f acc0 = {}, acc1 = {};
    #pragma unroll
    for (int kb = 0; kb < 8; kb += 2) {
      const v16h a0 = load_a_frag(arow, kb, half);
      acc0 = wmma16(a0, bfrag[kb], acc0);
      const v16h a1 = load_a_frag(arow, kb + 1, half);
      acc1 = wmma16(a1, bfrag[kb + 1], acc1);
    }

    float* orow = out + ((size_t)t * BATCH + brow0 + 8 * half) * HDIM + col;
    #pragma unroll
    for (int v = 0; v < 8; ++v) {
      const float th = FAST_TANH(acc0[v] + acc1[v] + xv[v]);
      orow[(size_t)v * HDIM] = th;                        // hs[t] (fire & forget)
      hnext[(v + 8 * half) * LDM + col] = (_Float16)th;   // h_t for next step
    }

    __syncthreads();  // single barrier: h_t visible, h_{t-1} buffer reusable
  }
}

// ---------------------------------------------------------------------------
extern "C" void kernel_launch(void* const* d_in, const int* in_sizes, int n_in,
                              void* d_out, int out_size, void* d_ws, size_t ws_size,
                              hipStream_t stream) {
  (void)in_sizes; (void)n_in; (void)out_size; (void)ws_size;
  const float* x    = (const float*)d_in[0];
  const float* W_ih = (const float*)d_in[1];
  const float* W_hh = (const float*)d_in[2];
  const float* b_ih = (const float*)d_in[3];
  const float* b_hh = (const float*)d_in[4];
  float* out = (float*)d_out;
  float* xpw = (float*)d_ws;  // needs T*B*H*4 = 128 MiB (fits 192 MB L2)

  rnn_xproj<<<(T_STEPS * BATCH) / 64, 512, 0, stream>>>(x, W_ih, b_ih, b_hh, xpw);
  rnn_recur<<<BATCH / 16, 512, 0, stream>>>(xpw, W_hh, out);
}
